// MaskedAutoEncoder_64845416235332
// MI455X (gfx1250) — matrix operand order repaired and checked by reading, not compiled, not run on hardware
//
#include <hip/hip_runtime.h>

// ---------------- problem constants ----------------
static constexpr int cB = 2;
static constexpr int cC = 5, cT = 16, cS = 64;
static constexpr int cTP = 2, cSP = 4;
static constexpr int cE = 768, cH = 12, cD = 64;
static constexpr int cTN = cT / cTP;                // 8
static constexpr int cSN = (cS / cSP) * (cS / cSP); // 256
static constexpr int cNPATCH = cTN * cSN;           // 2048
static constexpr int cNVIS = 512;
static constexpr int cENC_D = 8, cDEC_D = 2;
static constexpr int cNPIX = cC * cTP * cSP * cSP;  // 160

typedef unsigned short u16;
typedef __bf16 bf16;
typedef __attribute__((ext_vector_type(16))) __bf16 v16bf;
typedef __attribute__((ext_vector_type(8)))  float  v8f;
typedef __attribute__((ext_vector_type(4)))  int    v4i;

// CDNA5 async global->LDS copy path (ASYNCcnt-tracked DMA), guarded so the
// build never regresses if the toolchain lacks the builtins.
#if __has_builtin(__builtin_amdgcn_global_load_async_to_lds_b128) && \
    __has_builtin(__builtin_amdgcn_s_wait_asynccnt)
#define MAE_ASYNC_COPY 1
#else
#define MAE_ASYNC_COPY 0
#endif

#if MAE_ASYNC_COPY
typedef __attribute__((address_space(1))) v4i* gv4i_p;   // global (AS1) int4*
typedef __attribute__((address_space(3))) v4i* lv4i_p;   // LDS (AS3) int4*

__device__ __forceinline__ gv4i_p as_gv4i(const void* p) {
    // flat address of global memory == its AS1 address; integer round-trip
    // also drops const to match the builtin's parameter type
    return (gv4i_p)(unsigned long long)(size_t)p;
}
__device__ __forceinline__ lv4i_p as_lv4i(void* p) {
    return (lv4i_p)(v4i*)p;   // generic -> AS3 addrspacecast (pointer is LDS)
}
#endif

// ---------------- helpers ----------------
__device__ __forceinline__ u16 f2bf(float f) {
    unsigned u = __float_as_uint(f);
    u += 0x7FFFu + ((u >> 16) & 1u);   // round-to-nearest-even
    return (u16)(u >> 16);
}

// A-fragment (16x32 bf16, row-major source, row = lane&15).
// lanes 0-15: elems 0..15 -> K 0..15
// lanes 16-31: elems 0..15 -> K 16..31
__device__ __forceinline__ v16bf load_a_frag(const bf16* rowp, int lane) {
    const int kb = (lane < 16) ? 0 : 16;
    v16bf a;
#pragma unroll
    for (int i = 0; i < 16; ++i) a[i] = rowp[kb + i];
    return a;
}

// B-fragment (32x16 bf16) where the K dimension is contiguous in memory
// (pointer already offset by column*stride + kb, kb = (lane<16)?0:16)
__device__ __forceinline__ v16bf load_b_frag_contig(const bf16* p) {
    v16bf b;
#pragma unroll
    for (int i = 0; i < 16; ++i) b[i] = p[i];
    return b;
}

// B-fragment from a row-major [K x N] source (stride between K elements)
__device__ __forceinline__ v16bf load_b_frag_strided(const bf16* p, int stride) {
    v16bf b;
#pragma unroll
    for (int i = 0; i < 16; ++i) b[i] = p[i * stride];
    return b;
}

__device__ __forceinline__ v8f wmma_bf16(v16bf a, v16bf b, v8f c) {
    return __builtin_amdgcn_wmma_f32_16x16x32_bf16(false, a, false, b, (short)0, c, false, false);
}

// ---------------- tiny utility kernels ----------------
__global__ void f32_to_bf16_kernel(const float* __restrict__ s, u16* __restrict__ d, long long n) {
    long long id = (long long)blockIdx.x * 256 + threadIdx.x;
    if (id < n) d[id] = f2bf(s[id]);
}

__global__ void pos_kernel(float* __restrict__ pos) {
    int id = blockIdx.x * 256 + threadIdx.x;
    if (id >= cNPATCH * cE) return;
    int n = id / cE, e = id % cE;
    int a = n >> 8;        // temporal index (TN=8)
    int sidx = n & 255;    // spatial index (SN=256)
    int i = e >> 1;
    float denom = __powf(10000.0f, 2.0f * (float)i / (float)cE);
    float angT = (float)a / denom;
    float angS = (float)sidx / denom;
    float vT = (e & 1) ? __cosf(angT) : __sinf(angT);
    float vS = (e & 1) ? __cosf(angS) : __sinf(angS);
    pos[id] = vT + vS;
}

// patches[b, patch, e] = sum_k pix[k] * W[e,k] + bias[e] + pos[patch,e]
__global__ __launch_bounds__(256) void patch_embed_kernel(
    const float* __restrict__ X, const float* __restrict__ W,
    const float* __restrict__ bias, const float* __restrict__ pos,
    float* __restrict__ P)
{
    __shared__ float pix[cNPIX];
    const int patch = blockIdx.x, b = blockIdx.y, tid = threadIdx.x;
    const int a = patch >> 8, hh = (patch >> 4) & 15, ww = patch & 15;
    if (tid < cNPIX) {
        int c = tid >> 5, r = tid & 31;
        int tp = r >> 4, p = (r >> 2) & 3, q = r & 3;
        int t = a * cTP + tp, s1 = hh * cSP + p, s2 = ww * cSP + q;
        pix[tid] = X[(((size_t)b * cT + t) * cC + c) * (cS * cS) + s1 * cS + s2];
    }
    __syncthreads();
    for (int e = tid; e < cE; e += 256) {
        const float* w = W + (size_t)e * cNPIX;
        float acc = 0.f;
#pragma unroll 8
        for (int k = 0; k < cNPIX; ++k) acc += pix[k] * w[k];
        P[((size_t)b * cNPATCH + patch) * cE + e] = acc + bias[e] + pos[(size_t)patch * cE + e];
    }
}

__global__ void gather_vis_kernel(const float* __restrict__ P, const int* __restrict__ vis,
                                  float* __restrict__ Xo) {
    int id = blockIdx.x * 256 + threadIdx.x;
    if (id >= cB * cNVIS * cE) return;
    int e = id % cE; int rest = id / cE;
    int i = rest % cNVIS; int b = rest / cNVIS;
    Xo[id] = P[((size_t)b * cNPATCH + vis[i]) * cE + e];
}

__global__ void fill_full_kernel(const float* __restrict__ mask, const float* __restrict__ pos,
                                 float* __restrict__ full) {
    int id = blockIdx.x * 256 + threadIdx.x;
    if (id >= cB * cNPATCH * cE) return;
    int e = id % cE; int n = (id / cE) % cNPATCH;
    full[id] = mask[e] + pos[(size_t)n * cE + e];
}

__global__ void scatter_enc_kernel(const float* __restrict__ enc, const int* __restrict__ vis,
                                   const float* __restrict__ pos, float* __restrict__ full) {
    int id = blockIdx.x * 256 + threadIdx.x;
    if (id >= cB * cNVIS * cE) return;
    int e = id % cE; int rest = id / cE;
    int i = rest % cNVIS; int b = rest / cNVIS;
    int n = vis[i];
    full[((size_t)b * cNPATCH + n) * cE + e] =
        enc[((size_t)b * cNVIS + i) * cE + e] + pos[(size_t)n * cE + e];
}

__global__ void unpatch_kernel(const float* __restrict__ Tm, float* __restrict__ out) {
    int id = blockIdx.x * 256 + threadIdx.x;
    if (id >= cB * cT * cC * cS * cS) return;
    int s2 = id & 63, s1 = (id >> 6) & 63;
    int rest = id >> 12;
    int c = rest % cC; rest /= cC;
    int t = rest % cT; int b = rest / cT;
    int a = t >> 1, tp = t & 1, hh = s1 >> 2, p = s1 & 3, ww = s2 >> 2, q = s2 & 3;
    int patch = a * 256 + hh * 16 + ww;
    int k = c * 32 + tp * 16 + p * 4 + q;
    out[id] = Tm[((size_t)b * cNPATCH + patch) * cNPIX + k];
}

// ---------------- LayerNorm (f32 in -> bf16 out) ----------------
__global__ __launch_bounds__(256) void ln_kernel(
    const float* __restrict__ X, const float* __restrict__ g, const float* __restrict__ bt,
    u16* __restrict__ Y)
{
    __shared__ float red1[256];
    __shared__ float red2[256];
    const int row = blockIdx.x, tid = threadIdx.x;
    const float* x = X + (size_t)row * cE;
    float s = 0.f, ss = 0.f;
    for (int e = tid; e < cE; e += 256) { float v = x[e]; s += v; ss += v * v; }
    red1[tid] = s; red2[tid] = ss;
    __syncthreads();
    for (int w = 128; w > 0; w >>= 1) {
        if (tid < w) { red1[tid] += red1[tid + w]; red2[tid] += red2[tid + w]; }
        __syncthreads();
    }
    const float mean = red1[0] / (float)cE;
    const float var  = red2[0] / (float)cE - mean * mean;
    const float rstd = rsqrtf(var + 1e-5f);
    u16* y = Y + (size_t)row * cE;
    for (int e = tid; e < cE; e += 256)
        y[e] = f2bf((x[e] - mean) * rstd * g[e] + bt[e]);
}

// ---------------- split fused QKV (bf16 -> per-head q,k,v) ----------------
__global__ void split_qkv_kernel(const u16* __restrict__ QKV,
                                 u16* __restrict__ q, u16* __restrict__ k, u16* __restrict__ v,
                                 int Nseq) {
    long long id = (long long)blockIdx.x * 256 + threadIdx.x;
    long long tot = (long long)cB * Nseq * 3 * cE;
    if (id >= tot) return;
    int col = (int)(id % (3 * cE));
    long long row = id / (3 * cE);
    int b = (int)(row / Nseq), n = (int)(row % Nseq);
    int h = col / (3 * cD);
    int rr = col % (3 * cD);
    int dd = rr / 3, w = rr % 3;
    u16 val = QKV[id];
    size_t dst = (((size_t)b * cH + h) * Nseq + n) * cD + dd;
    if (w == 0) q[dst] = val; else if (w == 1) k[dst] = val; else v[dst] = val;
}

// ---------------- WMMA GEMM: C[M,N] = A[M,K](bf16) * B[K,N](bf16) + bias ----------------
// 4 waves per block, 64x64 block tile, 16x64 per wave, K step 32.
// Tiles are double-buffered; on CDNA5 they are staged by GLOBAL_LOAD_ASYNC_TO_LDS_B128
// (ASYNCcnt) so DMA overlaps the WMMA pipeline; fallback is synchronous b128 copies.
template<bool GELU, bool RESID, bool OUTBF>
__global__ __launch_bounds__(128) void gemm_kernel(
    const u16* __restrict__ A, const u16* __restrict__ B,
    const float* __restrict__ bias, const float* __restrict__ res,
    float* __restrict__ Cf, u16* __restrict__ Cb,
    int M, int N, int K)
{
    __shared__ __align__(16) u16 ldsA[2][64 * 32];   // [m][k]
    __shared__ __align__(16) u16 ldsB[2][32 * 64];   // [k][n] (natural layout -> DMA-copyable)
    const int tid = threadIdx.x;
    const int lane = tid & 31;
    const int wave = tid >> 5;
    const int m0 = blockIdx.y * 64;
    const int n0 = blockIdx.x * 64;
    const int nk = K >> 5;

    v8f acc[4];
#pragma unroll
    for (int j = 0; j < 4; ++j)
#pragma unroll
        for (int r = 0; r < 8; ++r) acc[j][r] = 0.f;

    // one tile = 4 async ops per thread (2x A b128, 2x B b128)
    auto issue_tile = [&](int ki, int bufi) {
        const int k0 = ki * 32;
#pragma unroll
        for (int ci = 0; ci < 2; ++ci) {
            const int chunk = tid * 2 + ci;
            const int rowA = chunk >> 2, kcA = (chunk & 3) << 3;   // 64x32 tile
            const int rowB = chunk >> 3, ncB = (chunk & 7) << 3;   // 32x64 tile
            const u16* asrc = &A[(size_t)(m0 + rowA) * K + k0 + kcA];
            const u16* bsrc = &B[(size_t)(k0 + rowB) * N + n0 + ncB];
            if (n0 + ncB >= N) bsrc = B;   // keep issue count constant; cols masked in epilogue
#if MAE_ASYNC_COPY
            __builtin_amdgcn_global_load_async_to_lds_b128(
                as_gv4i(asrc), as_lv4i(&ldsA[bufi][rowA * 32 + kcA]), 0, 0);
            __builtin_amdgcn_global_load_async_to_lds_b128(
                as_gv4i(bsrc), as_lv4i(&ldsB[bufi][rowB * 64 + ncB]), 0, 0);
#else
            *reinterpret_cast<uint4*>(&ldsA[bufi][rowA * 32 + kcA]) =
                *reinterpret_cast<const uint4*>(asrc);
            *reinterpret_cast<uint4*>(&ldsB[bufi][rowB * 64 + ncB]) =
                *reinterpret_cast<const uint4*>(bsrc);
#endif
        }
    };

    issue_tile(0, 0);
    if (nk > 1) issue_tile(1, 1);

    for (int ki = 0; ki < nk; ++ki) {
        const int bufi = ki & 1;
#if MAE_ASYNC_COPY
        // async loads complete in order: leaving <=4 outstanding means the
        // current tile's group is done and only the prefetched tile is in flight
        if (ki + 1 < nk) __builtin_amdgcn_s_wait_asynccnt(4);
        else             __builtin_amdgcn_s_wait_asynccnt(0);
#endif
        __syncthreads();

        const bf16* aRow = reinterpret_cast<const bf16*>(&ldsA[bufi][(wave * 16 + (lane & 15)) * 32]);
        v16bf af = load_a_frag(aRow, lane);
        const int kb2 = (lane < 16) ? 0 : 16;
#pragma unroll
        for (int j = 0; j < 4; ++j) {
            const bf16* bp = reinterpret_cast<const bf16*>(&ldsB[bufi][kb2 * 64 + j * 16 + (lane & 15)]);
            acc[j] = wmma_bf16(af, load_b_frag_strided(bp, 64), acc[j]);
        }
        __syncthreads();
        if (ki + 2 < nk) issue_tile(ki + 2, bufi);
    }

    // epilogue
#pragma unroll
    for (int j = 0; j < 4; ++j) {
        int col = n0 + j * 16 + (lane & 15);
        if (col >= N) continue;
#pragma unroll
        for (int r = 0; r < 8; ++r) {
            int row = m0 + wave * 16 + ((lane < 16) ? r : r + 8);
            float v = acc[j][r] + bias[col];
            if (RESID) v += res[(size_t)row * N + col];
            if (GELU)  v = 0.5f * v * (1.0f + erff(v * 0.70710678118654752f));
            if (OUTBF) Cb[(size_t)row * N + col] = f2bf(v);
            else       Cf[(size_t)row * N + col] = v;
        }
    }
}

// ---------------- attention: one wave handles (b, h, 16 query rows) ----------------
// energy = Q Kt ; att = softmax(energy)/sqrt(E) ; O = att V
// Single-pass flash: the WMMA C-layout (v8f elem r <-> row r / r+8 per lane half)
// matches the per-lane register layout of the row stats, so accumulator rescale
// is a pure per-component multiply.
__global__ __launch_bounds__(32) void attn_kernel(
    const u16* __restrict__ Q, const u16* __restrict__ Kk, const u16* __restrict__ V,
    u16* __restrict__ Out, int Nseq)
{
    __shared__ __align__(16) u16 Pld[16 * 32];
    const int lane = threadIdx.x;
    const int qt = blockIdx.x, h = blockIdx.y, b = blockIdx.z;
    const size_t bh = ((size_t)b * cH + h) * Nseq;
    const int q0 = qt * 16;
    const int c = lane & 15;
    const int kb2 = (lane < 16) ? 0 : 16;

    // Q fragments (16 rows x 64 feat) -> 2 A-frags of 16x32
    const bf16* qrow = reinterpret_cast<const bf16*>(&Q[(bh + q0 + c) * cD]);
    const v16bf aq0 = load_a_frag(qrow, lane);
    const v16bf aq1 = load_a_frag(qrow + 32, lane);

    float mrow[8], srow[8];
#pragma unroll
    for (int r = 0; r < 8; ++r) { mrow[r] = -3.0e38f; srow[r] = 0.f; }

    v8f oacc[4];
#pragma unroll
    for (int t = 0; t < 4; ++t)
#pragma unroll
        for (int r = 0; r < 8; ++r) oacc[t][r] = 0.f;

    for (int j0 = 0; j0 < Nseq; j0 += 32) {
        if (j0 + 64 < Nseq) {   // warm L2/WGP$ for upcoming K/V blocks
            __builtin_prefetch(&Kk[(bh + j0 + 64) * cD], 0, 0);
            __builtin_prefetch(&V[(bh + j0 + 64) * cD], 0, 0);
        }
        // ---- energy for 32 keys: two 16x16 tiles ----
        const bf16* kcol0 = reinterpret_cast<const bf16*>(&Kk[(bh + j0 + c) * cD + kb2]);
        const bf16* kcol1 = reinterpret_cast<const bf16*>(&Kk[(bh + j0 + 16 + c) * cD + kb2]);
        v8f e0, e1;
#pragma unroll
        for (int r = 0; r < 8; ++r) { e0[r] = 0.f; e1[r] = 0.f; }
        e0 = wmma_bf16(aq0, load_b_frag_contig(kcol0), e0);
        e0 = wmma_bf16(aq1, load_b_frag_contig(kcol0 + 32), e0);
        e1 = wmma_bf16(aq0, load_b_frag_contig(kcol1), e1);
        e1 = wmma_bf16(aq1, load_b_frag_contig(kcol1 + 32), e1);

        // ---- online softmax update + stage P (bf16) into LDS ----
        float sc[8];
#pragma unroll
        for (int r = 0; r < 8; ++r) {
            float tm = fmaxf(e0[r], e1[r]);
            tm = fmaxf(tm, __shfl_xor(tm, 8, 16));
            tm = fmaxf(tm, __shfl_xor(tm, 4, 16));
            tm = fmaxf(tm, __shfl_xor(tm, 2, 16));
            tm = fmaxf(tm, __shfl_xor(tm, 1, 16));
            const float nm = fmaxf(mrow[r], tm);
            const float p0 = __expf(e0[r] - nm);
            const float p1 = __expf(e1[r] - nm);
            const int row = (lane < 16) ? r : r + 8;
            Pld[row * 32 + c]      = f2bf(p0);
            Pld[row * 32 + 16 + c] = f2bf(p1);
            float ts = p0 + p1;
            ts += __shfl_xor(ts, 8, 16);
            ts += __shfl_xor(ts, 4, 16);
            ts += __shfl_xor(ts, 2, 16);
            ts += __shfl_xor(ts, 1, 16);
            const float s = __expf(mrow[r] - nm);
            srow[r] = srow[r] * s + ts;
            mrow[r] = nm;
            sc[r] = s;
        }
        // rescale accumulators (layout-aligned, pure per-component)
#pragma unroll
        for (int t = 0; t < 4; ++t)
#pragma unroll
            for (int r = 0; r < 8; ++r) oacc[t][r] *= sc[r];

        __syncthreads();
        const bf16* prow = reinterpret_cast<const bf16*>(&Pld[c * 32]);
        v16bf ap = load_a_frag(prow, lane);
#pragma unroll
        for (int t = 0; t < 4; ++t) {
            const bf16* vp = reinterpret_cast<const bf16*>(
                &V[(bh + j0 + kb2) * cD + t * 16 + c]);
            oacc[t] = wmma_bf16(ap, load_b_frag_strided(vp, cD), oacc[t]);
        }
        __syncthreads();
    }

    // ---- epilogue: scale by 1/(rowsum * sqrt(E)), write bf16 [B, N, E] ----
#pragma unroll
    for (int t = 0; t < 4; ++t) {
#pragma unroll
        for (int r = 0; r < 8; ++r) {
            int row = q0 + ((lane < 16) ? r : r + 8);
            float o = oacc[t][r] / (srow[r] * 27.712812921102035f); // sqrt(768)
            Out[((size_t)b * Nseq + row) * cE + h * cD + t * 16 + c] = f2bf(o);
        }
    }
}

// ---------------- host ----------------
struct NetP {
    const float *b1, *b2, *bproj, *bqkv, *ln1_b, *ln1_s, *ln2_b, *ln2_s, *w1, *w2, *wproj, *wqkv;
};

extern "C" void kernel_launch(void* const* d_in, const int* in_sizes, int n_in,
                              void* d_out, int out_size, void* d_ws, size_t ws_size,
                              hipStream_t stream)
{
    (void)in_sizes; (void)n_in; (void)out_size; (void)ws_size;
    // inputs flattened in jax pytree order (dict keys sorted)
    const float* x         = (const float*)d_in[0];
    const float* patch_w   = (const float*)d_in[1];
    const float* patch_b   = (const float*)d_in[2];
    const float* mask_tok  = (const float*)d_in[3];
    NetP enc = { (const float*)d_in[4],  (const float*)d_in[5],  (const float*)d_in[6],  (const float*)d_in[7],
                 (const float*)d_in[8],  (const float*)d_in[9],  (const float*)d_in[10], (const float*)d_in[11],
                 (const float*)d_in[12], (const float*)d_in[13], (const float*)d_in[14], (const float*)d_in[15] };
    NetP dcp = { (const float*)d_in[16], (const float*)d_in[17], (const float*)d_in[18], (const float*)d_in[19],
                 (const float*)d_in[20], (const float*)d_in[21], (const float*)d_in[22], (const float*)d_in[23],
                 (const float*)d_in[24], (const float*)d_in[25], (const float*)d_in[26], (const float*)d_in[27] };
    const float* pix_w   = (const float*)d_in[28];
    const float* pix_b   = (const float*)d_in[29];
    const int*   vis_idx = (const int*)d_in[30];

    const int ME = cB * cNVIS;     // 1024 encoder rows
    const int MD = cB * cNPATCH;   // 4096 decoder rows

    // ---- workspace carve ----
    char* wsp = (char*)d_ws;
    size_t off = 0;
    auto alloc = [&](size_t bytes) -> void* {
        void* p = wsp + off;
        off = (off + bytes + 255) & ~(size_t)255;
        return p;
    };
    float* pos     = (float*)alloc((size_t)cNPATCH * cE * 4);
    float* patches = (float*)alloc((size_t)cB * cNPATCH * cE * 4);
    float* xact    = (float*)alloc((size_t)MD * cE * 4);
    float* encSave = (float*)alloc((size_t)ME * cE * 4);
    float* tmpPix  = (float*)alloc((size_t)MD * cNPIX * 4);
    u16* yB   = (u16*)alloc((size_t)MD * cE * 2);
    u16* qkvB = (u16*)alloc((size_t)MD * 3 * cE * 2);
    u16* qB   = (u16*)alloc((size_t)MD * cE * 2);
    u16* kB   = (u16*)alloc((size_t)MD * cE * 2);
    u16* vB   = (u16*)alloc((size_t)MD * cE * 2);
    u16* attB = (u16*)alloc((size_t)MD * cE * 2);
    u16* hB   = (u16*)alloc((size_t)MD * 4 * cE * 2);
    u16* encWqkv  = (u16*)alloc((size_t)cENC_D * cE * 3 * cE * 2);
    u16* encWproj = (u16*)alloc((size_t)cENC_D * cE * cE * 2);
    u16* encW1    = (u16*)alloc((size_t)cENC_D * cE * 4 * cE * 2);
    u16* encW2    = (u16*)alloc((size_t)cENC_D * 4 * cE * cE * 2);
    u16* decWqkv  = (u16*)alloc((size_t)cDEC_D * cE * 3 * cE * 2);
    u16* decWproj = (u16*)alloc((size_t)cDEC_D * cE * cE * 2);
    u16* decW1    = (u16*)alloc((size_t)cDEC_D * cE * 4 * cE * 2);
    u16* decW2    = (u16*)alloc((size_t)cDEC_D * 4 * cE * cE * 2);
    u16* pixWB    = (u16*)alloc((size_t)cE * cNPIX * 2);

    auto conv = [&](const float* s, u16* d, long long n) {
        f32_to_bf16_kernel<<<(unsigned)((n + 255) / 256), 256, 0, stream>>>(s, d, n);
    };
    // weight conversions (once per launch, deterministic)
    conv(enc.wqkv,  encWqkv,  (long long)cENC_D * cE * 3 * cE);
    conv(enc.wproj, encWproj, (long long)cENC_D * cE * cE);
    conv(enc.w1,    encW1,    (long long)cENC_D * cE * 4 * cE);
    conv(enc.w2,    encW2,    (long long)cENC_D * 4 * cE * cE);
    conv(dcp.wqkv,  decWqkv,  (long long)cDEC_D * cE * 3 * cE);
    conv(dcp.wproj, decWproj, (long long)cDEC_D * cE * cE);
    conv(dcp.w1,    decW1,    (long long)cDEC_D * cE * 4 * cE);
    conv(dcp.w2,    decW2,    (long long)cDEC_D * 4 * cE * cE);
    conv(pix_w,     pixWB,    (long long)cE * cNPIX);

    // positions + patch embed + gather visible
    pos_kernel<<<(cNPATCH * cE + 255) / 256, 256, 0, stream>>>(pos);
    patch_embed_kernel<<<dim3(cNPATCH, cB), 256, 0, stream>>>(x, patch_w, patch_b, pos, patches);
    gather_vis_kernel<<<(cB * cNVIS * cE + 255) / 256, 256, 0, stream>>>(patches, vis_idx, xact);

    auto run_block = [&](float* X, int Nseq, int L, const NetP& P,
                         u16* Wqkv, u16* Wproj, u16* W1, u16* W2) {
        const int M = cB * Nseq;
        const float* ln1s = P.ln1_s + (size_t)L * cE;
        const float* ln1b = P.ln1_b + (size_t)L * cE;
        const float* ln2s = P.ln2_s + (size_t)L * cE;
        const float* ln2b = P.ln2_b + (size_t)L * cE;
        const float* bqkv = P.bqkv + (size_t)L * 3 * cE;
        const float* bprj = P.bproj + (size_t)L * cE;
        const float* b1   = P.b1 + (size_t)L * 4 * cE;
        const float* b2   = P.b2 + (size_t)L * cE;
        const u16* wqkv = Wqkv + (size_t)L * cE * 3 * cE;
        const u16* wprj = Wproj + (size_t)L * cE * cE;
        const u16* w1   = W1 + (size_t)L * cE * 4 * cE;
        const u16* w2   = W2 + (size_t)L * 4 * cE * cE;

        ln_kernel<<<M, 256, 0, stream>>>(X, ln1s, ln1b, yB);
        gemm_kernel<false, false, true><<<dim3(3 * cE / 64, M / 64), 128, 0, stream>>>(
            yB, wqkv, bqkv, nullptr, nullptr, qkvB, M, 3 * cE, cE);
        long long tot = (long long)M * 3 * cE;
        split_qkv_kernel<<<(unsigned)((tot + 255) / 256), 256, 0, stream>>>(qkvB, qB, kB, vB, Nseq);
        attn_kernel<<<dim3(Nseq / 16, cH, cB), 32, 0, stream>>>(qB, kB, vB, attB, Nseq);
        gemm_kernel<false, true, false><<<dim3(cE / 64, M / 64), 128, 0, stream>>>(
            attB, wprj, bprj, X, X, nullptr, M, cE, cE);
        ln_kernel<<<M, 256, 0, stream>>>(X, ln2s, ln2b, yB);
        gemm_kernel<true, false, true><<<dim3(4 * cE / 64, M / 64), 128, 0, stream>>>(
            yB, w1, b1, nullptr, nullptr, hB, M, 4 * cE, cE);
        gemm_kernel<false, true, false><<<dim3(cE / 64, M / 64), 128, 0, stream>>>(
            hB, w2, b2, X, X, nullptr, M, cE, 4 * cE);
    };

    // encoder
    for (int l = 0; l < cENC_D; ++l)
        run_block(xact, cNVIS, l, enc, encWqkv, encWproj, encW1, encW2);

    // build decoder input: mask token + pos, scatter encoded visible patches
    (void)hipMemcpyAsync(encSave, xact, (size_t)ME * cE * 4, hipMemcpyDeviceToDevice, stream);
    fill_full_kernel<<<(cB * cNPATCH * cE + 255) / 256, 256, 0, stream>>>(mask_tok, pos, xact);
    scatter_enc_kernel<<<(cB * cNVIS * cE + 255) / 256, 256, 0, stream>>>(encSave, vis_idx, pos, xact);

    // decoder
    for (int l = 0; l < cDEC_D; ++l)
        run_block(xact, cNPATCH, l, dcp, decWqkv, decWproj, decW1, decW2);

    // pixel head + un-patchify
    conv(xact, yB, (long long)MD * cE);
    gemm_kernel<false, false, false><<<dim3((cNPIX + 63) / 64, MD / 64), 128, 0, stream>>>(
        yB, pixWB, pix_b, nullptr, tmpPix, nullptr, MD, cNPIX, cE);
    unpatch_kernel<<<(cB * cT * cC * cS * cS + 255) / 256, 256, 0, stream>>>(tmpPix, (float*)d_out);
}
